// GoogleAttention_43739946943228
// MI455X (gfx1250) — matrix operand order repaired and checked by reading, not compiled, not run on hardware
//
#include <hip/hip_runtime.h>
#include <cmath>

// ---------------------------------------------------------------------------
// CDNA5 (gfx1250, wave32) WMMA bf16 pipeline for the attention block.
// All GEMMs are C = A * B^T with both operands K-contiguous (b128 loads per
// lane, layouts per ISA 7.12.2), software-pipelined so loads for k+1 are in
// flight while the WMMAs for k execute.
// ---------------------------------------------------------------------------

typedef __attribute__((ext_vector_type(16))) __bf16 v16bf;
typedef __attribute__((ext_vector_type(8)))  __bf16 v8bf;
typedef __attribute__((ext_vector_type(8)))  float  v8f;

__device__ __forceinline__ v8f wmma_bf16(v16bf a, v16bf b, v8f c) {
  // (neg_a, A, neg_b, B, c_mod, C, reuse_a, reuse_b)
  return __builtin_amdgcn_wmma_f32_16x16x32_bf16(false, a, false, b,
                                                 (short)0, c, false, false);
}

// A fragment, 16x32 bf16, A row-major [M,K]:
// lane (r=lane&15, half=lane>>4) reads two contiguous 8-element chunks at
// row r, k = kb+half*8 and kb+16+half*8  ->  two 16-byte loads.
__device__ __forceinline__ v16bf ld_a_frag(const __bf16* __restrict__ A,
                                           int lda, int m0, int kb, int lane) {
  const int r    = lane & 15;
  const int half = lane >> 4;
  const __bf16* p = A + (long)(m0 + r) * lda + kb + half * 8;
  v8bf lo = *(const v8bf*)p;
  v8bf hi = *(const v8bf*)(p + 16);
  return __builtin_shufflevector(lo, hi, 0, 1, 2, 3, 4, 5, 6, 7,
                                 8, 9, 10, 11, 12, 13, 14, 15);
}

// B fragment for logical B[k,n] = BT[n,k] (BT row-major [N,K]):
// lane (n=lane&15, khalf=lane>>4) reads two contiguous 8-element chunks at
// BT[n0+n, kb+khalf*8] and BT[n0+n, kb+16+khalf*8]  ->  two 16-byte loads.
__device__ __forceinline__ v16bf ld_bt_frag(const __bf16* __restrict__ BT,
                                            int ldb, int kb, int n0, int lane) {
  const int n     = lane & 15;
  const int khalf = lane >> 4;
  const __bf16* p = BT + (long)(n0 + n) * ldb + kb + khalf * 8;
  v8bf lo = *(const v8bf*)p;
  v8bf hi = *(const v8bf*)(p + 16);
  return __builtin_shufflevector(lo, hi, 0, 1, 2, 3, 4, 5, 6, 7,
                                 8, 9, 10, 11, 12, 13, 14, 15);
}

// C/D 16x16 f32: VGPR v -> M = v + 8*(lane/16), N = lane%16.
__device__ __forceinline__ void store_c_bf16(__bf16* __restrict__ C, int ldc,
                                             int m0, int n0, int lane, v8f acc) {
  const int n  = n0 + (lane & 15);
  const int mh = (lane >> 4) * 8;
#pragma unroll
  for (int v = 0; v < 8; ++v)
    C[(long)(m0 + mh + v) * ldc + n] = (__bf16)acc[v];
}

// ---------------------------------------------------------------------------
// Kernels
// ---------------------------------------------------------------------------

__global__ void k_f32_to_bf16(const float* __restrict__ in,
                              __bf16* __restrict__ out, long n) {
  long i      = (long)blockIdx.x * blockDim.x + threadIdx.x;
  long stride = (long)gridDim.x * blockDim.x;
  for (; i < n; i += stride) out[i] = (__bf16)in[i];
}

// in [R, D] f32  ->  out [D, R] bf16   (weight transpose + down-convert)
__global__ void k_cvt_transpose(const float* __restrict__ in,
                                __bf16* __restrict__ out, int R, int D) {
  long n      = (long)R * D;
  long i      = (long)blockIdx.x * blockDim.x + threadIdx.x;
  long stride = (long)gridDim.x * blockDim.x;
  for (; i < n; i += stride) {
    const long r = i / D;
    const long d = i - r * D;
    out[d * R + r] = (__bf16)in[i];
  }
}

// C[z] = A[z] * B[z]^T.  A [M,K] row-major, BT [N,K] row-major, C [M,N] bf16.
// Wave tile 16(M) x 64(N), 4 accumulators; block = 4 waves along M.
// Software-pipelined: fragments for kb+32 are loaded before the WMMAs of kb.
// Grid = (M/64, N/64, nbatch). No branches around WMMA (EXEC all-1s).
__global__ void k_gemm_abt(const __bf16* __restrict__ A, long sA,
                           const __bf16* __restrict__ BT, long sB,
                           __bf16* __restrict__ C, long sC,
                           int K, int lda, int ldb, int ldc) {
  const int lane = threadIdx.x & 31;
  const int wave = threadIdx.x >> 5;
  const __bf16* Ab = A  + (long)blockIdx.z * sA;
  const __bf16* Bb = BT + (long)blockIdx.z * sB;
  __bf16*       Cb = C  + (long)blockIdx.z * sC;
  const int m0 = (blockIdx.x * 4 + wave) * 16;
  const int n0 = blockIdx.y * 64;
  v8f acc0 = {}, acc1 = {}, acc2 = {}, acc3 = {};

  // prologue: fragments for kb = 0
  v16bf a  = ld_a_frag(Ab, lda, m0, 0, lane);
  v16bf b0 = ld_bt_frag(Bb, ldb, 0, n0,      lane);
  v16bf b1 = ld_bt_frag(Bb, ldb, 0, n0 + 16, lane);
  v16bf b2 = ld_bt_frag(Bb, ldb, 0, n0 + 32, lane);
  v16bf b3 = ld_bt_frag(Bb, ldb, 0, n0 + 48, lane);

  for (int kb = 0; kb < K - 32; kb += 32) {
    const int kn = kb + 32;
    __builtin_prefetch(Ab + (long)(m0 + (lane & 15)) * lda + kn + 32, 0, 3);
    // issue next-iteration loads first so they overlap this step's WMMAs
    v16bf an  = ld_a_frag(Ab, lda, m0, kn, lane);
    v16bf bn0 = ld_bt_frag(Bb, ldb, kn, n0,      lane);
    v16bf bn1 = ld_bt_frag(Bb, ldb, kn, n0 + 16, lane);
    v16bf bn2 = ld_bt_frag(Bb, ldb, kn, n0 + 32, lane);
    v16bf bn3 = ld_bt_frag(Bb, ldb, kn, n0 + 48, lane);
    acc0 = wmma_bf16(a, b0, acc0);
    acc1 = wmma_bf16(a, b1, acc1);
    acc2 = wmma_bf16(a, b2, acc2);
    acc3 = wmma_bf16(a, b3, acc3);
    a = an; b0 = bn0; b1 = bn1; b2 = bn2; b3 = bn3;
  }
  // epilogue: last k-step
  acc0 = wmma_bf16(a, b0, acc0);
  acc1 = wmma_bf16(a, b1, acc1);
  acc2 = wmma_bf16(a, b2, acc2);
  acc3 = wmma_bf16(a, b3, acc3);

  store_c_bf16(Cb, ldc, m0, n0,      lane, acc0);
  store_c_bf16(Cb, ldc, m0, n0 + 16, lane, acc1);
  store_c_bf16(Cb, ldc, m0, n0 + 32, lane, acc2);
  store_c_bf16(Cb, ldc, m0, n0 + 48, lane, acc3);
}

// out = gamma * (O * WvT^T) + x   (f32 epilogue, same pipelined GEMM)
__global__ void k_final(const __bf16* __restrict__ O,
                        const __bf16* __restrict__ WvT,
                        const float* __restrict__ x,
                        const float* __restrict__ gamma,
                        float* __restrict__ out) {
  const int lane = threadIdx.x & 31;
  const int wave = threadIdx.x >> 5;
  const int m0 = (blockIdx.x * 4 + wave) * 16;   // rows of [32768, 512]
  const int n0 = blockIdx.y * 64;
  v8f acc0 = {}, acc1 = {}, acc2 = {}, acc3 = {};

  v16bf a  = ld_a_frag(O, 256, m0, 0, lane);
  v16bf b0 = ld_bt_frag(WvT, 256, 0, n0,      lane);
  v16bf b1 = ld_bt_frag(WvT, 256, 0, n0 + 16, lane);
  v16bf b2 = ld_bt_frag(WvT, 256, 0, n0 + 32, lane);
  v16bf b3 = ld_bt_frag(WvT, 256, 0, n0 + 48, lane);
#pragma unroll
  for (int kb = 0; kb < 256 - 32; kb += 32) {
    const int kn = kb + 32;
    v16bf an  = ld_a_frag(O, 256, m0, kn, lane);
    v16bf bn0 = ld_bt_frag(WvT, 256, kn, n0,      lane);
    v16bf bn1 = ld_bt_frag(WvT, 256, kn, n0 + 16, lane);
    v16bf bn2 = ld_bt_frag(WvT, 256, kn, n0 + 32, lane);
    v16bf bn3 = ld_bt_frag(WvT, 256, kn, n0 + 48, lane);
    acc0 = wmma_bf16(a, b0, acc0);
    acc1 = wmma_bf16(a, b1, acc1);
    acc2 = wmma_bf16(a, b2, acc2);
    acc3 = wmma_bf16(a, b3, acc3);
    a = an; b0 = bn0; b1 = bn1; b2 = bn2; b3 = bn3;
  }
  acc0 = wmma_bf16(a, b0, acc0);
  acc1 = wmma_bf16(a, b1, acc1);
  acc2 = wmma_bf16(a, b2, acc2);
  acc3 = wmma_bf16(a, b3, acc3);

  const float g  = gamma[0];
  const int   nb = lane & 15;
  const int   mh = (lane >> 4) * 8;
  const v8f accs[4] = {acc0, acc1, acc2, acc3};
#pragma unroll
  for (int j = 0; j < 4; ++j) {
    const int n = n0 + 16 * j + nb;
#pragma unroll
    for (int v = 0; v < 8; ++v) {
      const long idx = (long)(m0 + mh + v) * 512 + n;
      out[idx] = g * accs[j][v] + x[idx];
    }
  }
}

// 2x2 max pool, stride 2, NHWC (64x64 -> 32x32). in [B*4096, Cc] -> out [B*1024, Cc]
__global__ void k_maxpool2(const __bf16* __restrict__ in,
                           __bf16* __restrict__ out, int Cc) {
  const long n = 8192L * Cc;
  long i       = (long)blockIdx.x * blockDim.x + threadIdx.x;
  long stride  = (long)gridDim.x * blockDim.x;
  for (; i < n; i += stride) {
    const long ro = i / Cc;
    const int  c  = (int)(i - ro * Cc);
    const int  b  = (int)(ro >> 10);
    const int  p  = (int)(ro & 1023);
    const int  ph = p >> 5, pw = p & 31;
    const long ib = ((long)b * 4096 + ph * 128 + pw * 2) * Cc + c;
    float m = (float)in[ib];
    m = fmaxf(m, (float)in[ib + Cc]);
    m = fmaxf(m, (float)in[ib + 64L * Cc]);
    m = fmaxf(m, (float)in[ib + 65L * Cc]);
    out[i] = (__bf16)m;
  }
}

// Same pool but output transposed per batch: out[b][Cc][1024] (K-contiguous
// over pooled pixels for the attention GEMM's B operand).
__global__ void k_maxpool2_t(const __bf16* __restrict__ in,
                             __bf16* __restrict__ out, int Cc) {
  const long n = 8192L * Cc;
  long i       = (long)blockIdx.x * blockDim.x + threadIdx.x;
  long stride  = (long)gridDim.x * blockDim.x;
  for (; i < n; i += stride) {
    const int  p  = (int)(i & 1023);          // pooled pixel
    const long bc = i >> 10;                  // b * Cc + c
    const int  c  = (int)(bc % Cc);
    const int  b  = (int)(bc / Cc);
    const int  ph = p >> 5, pw = p & 31;
    const long ib = ((long)b * 4096 + ph * 128 + pw * 2) * Cc + c;
    float m = (float)in[ib];
    m = fmaxf(m, (float)in[ib + Cc]);
    m = fmaxf(m, (float)in[ib + 64L * Cc]);
    m = fmaxf(m, (float)in[ib + 65L * Cc]);
    out[i] = (__bf16)m;                       // out[((b*Cc)+c)*1024 + p]
  }
}

// sT: [8][4096][1024] bf16. Softmax over the 4096 (row) dim for each (b, col):
// one thread per (b, col); lanes sweep consecutive columns -> coalesced.
__global__ void k_softmax_cols(__bf16* __restrict__ sT) {
  const int t   = blockIdx.x * blockDim.x + threadIdx.x;   // 0..8191
  const int b   = t >> 10;
  const int col = t & 1023;
  __bf16* base = sT + (long)b * 4096 * 1024 + col;
  float mx = -3.4e38f;
  for (int m = 0; m < 4096; ++m)
    mx = fmaxf(mx, (float)base[(long)m * 1024]);
  float sum = 0.f;
  for (int m = 0; m < 4096; ++m)
    sum += __expf((float)base[(long)m * 1024] - mx);
  const float inv = 1.f / sum;
  for (int m = 0; m < 4096; ++m) {
    const long idx = (long)m * 1024;
    base[idx] = (__bf16)(__expf((float)base[idx] - mx) * inv);
  }
}

// ---------------------------------------------------------------------------
// Launch
// ---------------------------------------------------------------------------

extern "C" void kernel_launch(void* const* d_in, const int* in_sizes, int n_in,
                              void* d_out, int out_size, void* d_ws, size_t ws_size,
                              hipStream_t stream) {
  const float* x     = (const float*)d_in[0];
  const float* Wf    = (const float*)d_in[1];
  const float* Wg    = (const float*)d_in[2];
  const float* Wh    = (const float*)d_in[3];
  const float* Wv    = (const float*)d_in[4];
  const float* gamma = (const float*)d_in[5];
  float*       out   = (float*)d_out;

  constexpr long Bn = 8, C = 512, c1 = 64, c2 = 256;
  constexpr long n1 = 1024, n2 = 4096, M = Bn * n2;   // 32768 pixel rows

  char* p = (char*)d_ws;
  auto carve = [&](long bytes) { char* r = p; p += (bytes + 255) & ~255L; return r; };
  __bf16* xb    = (__bf16*)carve(M * C * 2);          // x bf16 [32768,512]
  __bf16* WfT   = (__bf16*)carve(c1 * C * 2);         // [64,512]
  __bf16* WgT   = (__bf16*)carve(c1 * C * 2);         // [64,512]
  __bf16* WhT   = (__bf16*)carve(c2 * C * 2);         // [256,512]
  __bf16* WvT   = (__bf16*)carve(C * c2 * 2);         // [512,256]
  __bf16* gf    = (__bf16*)carve(Bn * n2 * c1 * 2);   // g [b,4096,64]
  __bf16* ffull = (__bf16*)carve(M * c1 * 2);         // f pre-pool
  __bf16* ff    = (__bf16*)carve(Bn * n1 * c1 * 2);   // f pooled [b,1024,64]
  __bf16* hfull = (__bf16*)carve(M * c2 * 2);         // h pre-pool
  __bf16* hfT   = (__bf16*)carve(Bn * c2 * n1 * 2);   // h pooled^T [b,256,1024]
  __bf16* sT    = (__bf16*)carve(Bn * n2 * n1 * 2);   // s^T / beta^T [b,4096,1024]
  __bf16* obuf  = (__bf16*)carve(Bn * n2 * c2 * 2);   // o [b,4096,256]
  (void)ws_size; (void)in_sizes; (void)n_in; (void)out_size;

  const dim3 blk(128);  // 4 wave32s

  // 1) convert x; transpose+convert weights (so GEMM B operands are [N,K])
  k_f32_to_bf16 <<<4096, 256, 0, stream>>>(x,  xb, M * C);
  k_cvt_transpose<<<128, 256, 0, stream>>>(Wf, WfT, (int)C, (int)c1);
  k_cvt_transpose<<<128, 256, 0, stream>>>(Wg, WgT, (int)C, (int)c1);
  k_cvt_transpose<<<512, 256, 0, stream>>>(Wh, WhT, (int)C, (int)c2);
  k_cvt_transpose<<<512, 256, 0, stream>>>(Wv, WvT, (int)c2, (int)C);

  // 2) projections: g/f/h = xb * W^T  (WMMA)
  k_gemm_abt<<<dim3(M / 64, c1 / 64, 1), blk, 0, stream>>>(
      xb, 0, WgT, 0, gf,    0, (int)C, (int)C, (int)C, (int)c1);
  k_gemm_abt<<<dim3(M / 64, c1 / 64, 1), blk, 0, stream>>>(
      xb, 0, WfT, 0, ffull, 0, (int)C, (int)C, (int)C, (int)c1);
  k_gemm_abt<<<dim3(M / 64, c2 / 64, 1), blk, 0, stream>>>(
      xb, 0, WhT, 0, hfull, 0, (int)C, (int)C, (int)C, (int)c2);

  // 3) 2x2 max pools (h pooled directly into transposed layout)
  k_maxpool2  <<<2048, 256, 0, stream>>>(ffull, ff,  (int)c1);
  k_maxpool2_t<<<8192, 256, 0, stream>>>(hfull, hfT, (int)c2);

  // 4) sT[b] = gf[b] * ff[b]^T : [4096,64] x [64,1024] -> [4096,1024] (WMMA)
  k_gemm_abt<<<dim3(n2 / 64, n1 / 64, Bn), blk, 0, stream>>>(
      gf, n2 * c1, ff, n1 * c1, sT, n2 * n1, (int)c1, (int)c1, (int)c1, (int)n1);

  // 5) softmax over the n2 axis == column softmax of sT, in place
  k_softmax_cols<<<32, 256, 0, stream>>>(sT);

  // 6) o[b] = betaT[b] * hfT[b]^T : [4096,1024] x [1024,256] (WMMA)
  k_gemm_abt<<<dim3(n2 / 64, c2 / 64, Bn), blk, 0, stream>>>(
      sT, n2 * n1, hfT, c2 * n1, obuf, n2 * c2, (int)n1, (int)n1, (int)n1, (int)c2);

  // 7) out = gamma * (o * Wv) + x  (WMMA + f32 residual epilogue)
  k_final<<<dim3(M / 64, C / 64, 1), blk, 0, stream>>>(obuf, WvT, x, gamma, out);
}
